// SVTransformerBPETokenV7Fast_25555055411786
// MI455X (gfx1250) — matrix operand order repaired and checked by reading, not compiled, not run on hardware
//
#include <hip/hip_runtime.h>
#include <hip/hip_bf16.h>
#include <math.h>

// ---------------------------------------------------------------------------
// Problem constants (from reference): V=2048, MAXLEN=64, H=8, D=256, W=4096
// ---------------------------------------------------------------------------
#define V_      2048
#define MAXLEN_ 64
#define H_      8
#define D_      256
#define W_      4096
#define OUT3_   6144   // 3*H*D
#define PAD_    0

typedef __bf16 bf16;
typedef __attribute__((ext_vector_type(16))) __bf16 v16bf;
typedef __attribute__((ext_vector_type(8)))  __bf16 v8bf;
typedef __attribute__((ext_vector_type(8)))  float  v8f;

#define WMMA_BF16(A, B, C) \
  __builtin_amdgcn_wmma_f32_16x16x32_bf16(false, (A), false, (B), (short)0, (C), false, false)

// ---------------------------------------------------------------------------
// WMMA fragment loaders (wave32, CDNA5 ISA 7.12.2 layouts)
// A 16x32 bf16: lane = m + 16*g; halves 0..7 = K[8g..8g+7], 8..15 = K[16+8g..23+8g]
// B 32x16 bf16 sourced from [N,K] row-major: lane = n + 16*g; halves 0..7 = K[8g..8g+7], 8..15 = K[16+8g..23+8g]
// ---------------------------------------------------------------------------
__device__ __forceinline__ v16bf load_fragA(const bf16* __restrict__ p0, int ldk, int lane) {
  int r = lane & 15, g = lane >> 4;
  const bf16* p = p0 + (size_t)r * ldk + 8 * g;
  v8bf lo = *reinterpret_cast<const v8bf*>(p);
  v8bf hi = *reinterpret_cast<const v8bf*>(p + 16);
  return __builtin_shufflevector(lo, hi, 0,1,2,3,4,5,6,7,8,9,10,11,12,13,14,15);
}

__device__ __forceinline__ v16bf load_fragB(const bf16* __restrict__ p0, int ldk, int lane) {
  int n = lane & 15, g = lane >> 4;
  const bf16* p = p0 + (size_t)n * ldk + 8 * g;
  v8bf lo = *reinterpret_cast<const v8bf*>(p);
  v8bf hi = *reinterpret_cast<const v8bf*>(p + 16);
  return __builtin_shufflevector(lo, hi, 0,1,2,3,4,5,6,7,8,9,10,11,12,13,14,15);
}

// ---------------------------------------------------------------------------
// Generic batched GEMM: C[z][m][n] = scale * sum_k A[z][m][k]*B[z][n][k]
// (optionally + A2/B2 second product), K multiple of 32.
// Block: 256 threads = 8 waves, block tile 128(M) x 64(N), wave tile 32x32.
// Optional bf16 mirror of C; optional "column 0 -> -1e9" PAD mask.
// ---------------------------------------------------------------------------
__global__ __launch_bounds__(256) void wmma_gemm_nt(
    const bf16* __restrict__ A,  int lda, long long aZ,
    const bf16* __restrict__ B,  int ldb, long long bZ,
    const bf16* __restrict__ A2, const bf16* __restrict__ B2,
    float* __restrict__ C, int ldc, long long cZ,
    bf16* __restrict__ Cbf,
    int K, float scale, int maskCol0)
{
  const int z = blockIdx.z;
  const bf16* Az = A + (size_t)z * aZ;
  const bf16* Bz = B + (size_t)z * bZ;
  float*      Cz = C + (size_t)z * cZ;

  const int lane = threadIdx.x & 31;
  const int wv   = threadIdx.x >> 5;       // 0..7
  const int wm   = wv >> 1;                // 0..3
  const int wn   = wv & 1;                 // 0..1
  const int row0 = blockIdx.x * 128 + wm * 32;
  const int col0 = blockIdx.y * 64  + wn * 32;

  v8f acc[2][2] = {};

  const int npass = (A2 != nullptr) ? 2 : 1;
  for (int pass = 0; pass < npass; ++pass) {
    const bf16* Ap = pass ? (A2 + (size_t)z * aZ) : Az;
    const bf16* Bp = pass ? (B2 + (size_t)z * bZ) : Bz;
    const bf16* a_base = Ap + (size_t)row0 * lda;
    const bf16* b_base = Bp + (size_t)col0 * ldb;
    #pragma unroll 2
    for (int k0 = 0; k0 < K; k0 += 32) {
      v16bf a0 = load_fragA(a_base + k0, lda, lane);
      v16bf a1 = load_fragA(a_base + (size_t)16 * lda + k0, lda, lane);
      v16bf b0 = load_fragB(b_base + k0, ldb, lane);
      v16bf b1 = load_fragB(b_base + (size_t)16 * ldb + k0, ldb, lane);
      acc[0][0] = WMMA_BF16(a0, b0, acc[0][0]);
      acc[0][1] = WMMA_BF16(a0, b1, acc[0][1]);
      acc[1][0] = WMMA_BF16(a1, b0, acc[1][0]);
      acc[1][1] = WMMA_BF16(a1, b1, acc[1][1]);
    }
  }

  // Epilogue: C/D layout — lane = n + 16*g, VGPR r holds M = r + 8*g
  const int n16 = lane & 15;
  const int g   = lane >> 4;
  #pragma unroll
  for (int tm = 0; tm < 2; ++tm)
    #pragma unroll
    for (int tn = 0; tn < 2; ++tn)
      #pragma unroll
      for (int rr = 0; rr < 8; ++rr) {
        int m = row0 + tm * 16 + rr + 8 * g;
        int n = col0 + tn * 16 + n16;
        float v = acc[tm][tn][rr] * scale;
        if (maskCol0 && n == 0) v = -1.0e9f;   // char_qk[:, :, PAD] = -1e9
        size_t off = (size_t)m * ldc + n;
        Cz[off] = v;
        if (Cbf) Cbf[(size_t)z * cZ + off] = (bf16)v;
      }
}

// ---------------------------------------------------------------------------
// fp32 -> bf16 conversion
// ---------------------------------------------------------------------------
__global__ __launch_bounds__(256) void k_f32_to_bf16(const float* __restrict__ src,
                                                     bf16* __restrict__ dst, int n) {
  int i = blockIdx.x * 256 + threadIdx.x;
  if (i < n) dst[i] = (bf16)src[i];
}

// ---------------------------------------------------------------------------
// Sinusoidal positional table, zero-padded to 128 rows so it is M-tile aligned
// for the WMMA GEMM (rows 64..127 = 0 -> pos_qkv rows 64..127 = 0, harmless).
// ---------------------------------------------------------------------------
__global__ __launch_bounds__(256) void k_pos(bf16* __restrict__ posbf) {
  int i = blockIdx.x * 256 + threadIdx.x;       // 0..128*256-1
  int l = i >> 8, d = i & 255;
  float v = 0.0f;
  if (l < MAXLEN_) {
    int k = d >> 1;
    float freq = __powf(10000.0f, -(float)k / 128.0f);
    float ph   = (float)l * freq;
    v = (d & 1) ? __cosf(ph) : __sinf(ph);
  }
  posbf[i] = (bf16)v;
}

// ---------------------------------------------------------------------------
// Per-(word, head) attention: gather sim from the L2-resident char_qk table,
// softmax rows, mask-average rows into pooled, then out = pooled @ (V + posV).
// Block = 64 threads (2 waves, wave32); L read from device scalar (<=64).
// ---------------------------------------------------------------------------
__global__ __launch_bounds__(64) void k_attn(
    const int*   __restrict__ code,       // [W][MAXLEN]
    const float* __restrict__ charQK,     // [H][V][V]
    const float* __restrict__ charPosQK,  // [H][V][64]
    const float* __restrict__ posQK,      // [H][128][64]
    const float* __restrict__ qkvF,       // [V][6144] fp32
    const float* __restrict__ pqkvF,      // [128][6144] fp32
    float* __restrict__ outWS,            // [W][H][D]
    const int* __restrict__ maxlenp)
{
  __shared__ int   s_idx[MAXLEN_];
  __shared__ float s_sim[MAXLEN_][MAXLEN_ + 1];
  __shared__ float s_pool[MAXLEN_];
  __shared__ int   s_n;

  const int w = blockIdx.x;
  const int h = blockIdx.y;
  const int tid = threadIdx.x;
  int L = *maxlenp; if (L > MAXLEN_) L = MAXLEN_; if (L < 1) L = 1;

  if (tid < MAXLEN_) s_idx[tid] = (tid < L) ? code[(size_t)w * MAXLEN_ + tid] : PAD_;
  if (tid == 0) {
    int c = 0;
    for (int l = 0; l < L; ++l) c += (code[(size_t)w * MAXLEN_ + l] != PAD_);
    s_n = (c > 0) ? c : 1;
  }
  __syncthreads();

  // sim[i][j] = char_qk[h, idx_i, idx_j] + char_pos_qk[h, idx_i, j] + pos_qk[h, i, j]
  for (int e = tid; e < L * L; e += 64) {
    int i = e / L, j = e - i * L;
    int ii = s_idx[i], jj = s_idx[j];
    float v = charQK[((size_t)h * V_ + ii) * V_ + jj]
            + charPosQK[((size_t)h * V_ + ii) * MAXLEN_ + j]
            + posQK[((size_t)h * 128 + i) * MAXLEN_ + j];
    s_sim[i][j] = v;
  }
  __syncthreads();

  // row softmax; rows with PAD token contribute 0 to the pooled average
  if (tid < L) {
    if (s_idx[tid] != PAD_) {
      float mx = -3.4e38f;
      for (int j = 0; j < L; ++j) mx = fmaxf(mx, s_sim[tid][j]);
      float ssum = 0.0f;
      for (int j = 0; j < L; ++j) { float e = __expf(s_sim[tid][j] - mx); s_sim[tid][j] = e; ssum += e; }
      float inv = 1.0f / ssum;
      for (int j = 0; j < L; ++j) s_sim[tid][j] *= inv;
    } else {
      for (int j = 0; j < L; ++j) s_sim[tid][j] = 0.0f;
    }
  }
  __syncthreads();

  if (tid < L) {
    float p = 0.0f;
    for (int i = 0; i < L; ++i) p += s_sim[i][tid];
    s_pool[tid] = p / (float)s_n;
  }
  __syncthreads();

  // out[d] = sum_j pooled[j] * (char_v[idx_j,h,d] + pos_v[h,j,d]); s=2 offset = 4096
  const size_t vofs = (size_t)2 * 2048 + (size_t)h * D_;
  #pragma unroll
  for (int q = 0; q < 4; ++q) {
    int d = tid + q * 64;
    float acc = 0.0f;
    for (int j = 0; j < L; ++j) {
      float vv = qkvF[(size_t)s_idx[j] * OUT3_ + vofs + d]
               + pqkvF[(size_t)j * OUT3_ + vofs + d];
      acc += s_pool[j] * vv;
    }
    outWS[((size_t)w * H_ + h) * D_ + d] = acc;
  }
}

// ---------------------------------------------------------------------------
// LayerNorm over D per (w,h) + tiny MLP over heads: y[w,d].
// 256 threads = 8 waves; wave h reduces its head with wave32 shuffles.
// ---------------------------------------------------------------------------
__global__ __launch_bounds__(256) void k_head(
    const float* __restrict__ outWS,   // [W][H][D]
    const float* __restrict__ gamma, const float* __restrict__ beta,
    const float* __restrict__ w1, const float* __restrict__ b1,
    const float* __restrict__ w2, const float* __restrict__ b2,
    float* __restrict__ y)             // [W][D]
{
  __shared__ float s_mu[H_], s_rs[H_];
  const int w    = blockIdx.x;
  const int tid  = threadIdx.x;
  const int hw   = tid >> 5;           // wave id == head
  const int lane = tid & 31;
  const float* base = outWS + (size_t)w * H_ * D_;

  {
    float s = 0.0f, s2 = 0.0f;
    for (int i = lane; i < D_; i += 32) {
      float v = base[hw * D_ + i];
      s += v; s2 += v * v;
    }
    #pragma unroll
    for (int o = 16; o; o >>= 1) {
      s  += __shfl_xor(s,  o, 32);
      s2 += __shfl_xor(s2, o, 32);
    }
    if (lane == 0) {
      float mu = s * (1.0f / D_);
      s_mu[hw] = mu;
      s_rs[hw] = rsqrtf(s2 * (1.0f / D_) - mu * mu + 1e-5f);
    }
  }
  __syncthreads();

  const int d = tid;
  float f[H_];
  #pragma unroll
  for (int hh = 0; hh < H_; ++hh)
    f[hh] = (base[hh * D_ + d] - s_mu[hh]) * s_rs[hh] * gamma[d] + beta[d];

  float acc = b2[0];
  #pragma unroll
  for (int k = 0; k < 16; ++k) {
    float t = b1[k];
    #pragma unroll
    for (int hh = 0; hh < H_; ++hh) t += w1[k * H_ + hh] * f[hh];
    acc += w2[k] * tanhf(t);
  }
  y[(size_t)w * D_ + d] = acc;
}

// ---------------------------------------------------------------------------
// Host-side orchestration (graph-capture safe: only kernel launches on stream)
// ---------------------------------------------------------------------------
extern "C" void kernel_launch(void* const* d_in, const int* in_sizes, int n_in,
                              void* d_out, int out_size, void* d_ws, size_t ws_size,
                              hipStream_t stream) {
  (void)in_sizes; (void)n_in; (void)out_size; (void)ws_size;

  const int*   char_code = (const int*)  d_in[0];
  const float* char_emb  = (const float*)d_in[1];
  const float* w_char    = (const float*)d_in[2];
  const float* w_pos     = (const float*)d_in[3];
  const float* ln_gamma  = (const float*)d_in[4];
  const float* ln_beta   = (const float*)d_in[5];
  const float* w1        = (const float*)d_in[6];
  const float* b1        = (const float*)d_in[7];
  const float* w2        = (const float*)d_in[8];
  const float* b2        = (const float*)d_in[9];
  const int*   maxlen    = (const int*)  d_in[10];
  float* out = (float*)d_out;

  // Workspace carve-out (~260 MB total; char_qk table 134 MB sits in L2)
  char* wsb = (char*)d_ws;
  size_t off = 0;
  auto carve = [&](size_t bytes) -> void* {
    void* p = wsb + off;
    off += (bytes + 255) & ~(size_t)255;
    return p;
  };
  bf16*  Ebf    = (bf16*) carve((size_t)V_ * D_ * 2);
  bf16*  Wcbf   = (bf16*) carve((size_t)OUT3_ * D_ * 2);
  bf16*  Wpbf   = (bf16*) carve((size_t)OUT3_ * D_ * 2);
  bf16*  Pbf    = (bf16*) carve((size_t)128 * D_ * 2);
  float* QKVf   = (float*)carve((size_t)V_ * OUT3_ * 4);
  bf16*  QKVbf  = (bf16*) carve((size_t)V_ * OUT3_ * 2);
  float* PQKVf  = (float*)carve((size_t)128 * OUT3_ * 4);
  bf16*  PQKVbf = (bf16*) carve((size_t)128 * OUT3_ * 2);
  float* CQK    = (float*)carve((size_t)H_ * V_ * V_ * 4);
  float* CPQK   = (float*)carve((size_t)H_ * V_ * MAXLEN_ * 4);
  float* PQK    = (float*)carve((size_t)H_ * 128 * MAXLEN_ * 4);
  float* OUTw   = (float*)carve((size_t)W_ * H_ * D_ * 4);

  const float invTemp = 1.0f / 16.0f;    // 1/sqrt(D)

  // --- operand conversions + positional table ---
  k_f32_to_bf16<<<(V_ * D_ + 255) / 256, 256, 0, stream>>>(char_emb, Ebf, V_ * D_);
  k_f32_to_bf16<<<(OUT3_ * D_ + 255) / 256, 256, 0, stream>>>(w_char, Wcbf, OUT3_ * D_);
  k_f32_to_bf16<<<(OUT3_ * D_ + 255) / 256, 256, 0, stream>>>(w_pos, Wpbf, OUT3_ * D_);
  k_pos<<<(128 * D_) / 256, 256, 0, stream>>>(Pbf);

  // --- char_qkv = char_emb @ Wc^T  (2048 x 6144, K=256), fp32 + bf16 mirror ---
  wmma_gemm_nt<<<dim3(V_ / 128, OUT3_ / 64, 1), 256, 0, stream>>>(
      Ebf, D_, 0, Wcbf, D_, 0, nullptr, nullptr,
      QKVf, OUT3_, 0, QKVbf, D_, 1.0f, 0);

  // --- pos_qkv = pos @ Wp^T  (128(pad) x 6144, K=256) ---
  wmma_gemm_nt<<<dim3(1, OUT3_ / 64, 1), 256, 0, stream>>>(
      Pbf, D_, 0, Wpbf, D_, 0, nullptr, nullptr,
      PQKVf, OUT3_, 0, PQKVbf, D_, 1.0f, 0);

  // --- char_qk[h] = Q_h @ K_h^T / 16  (8 x 2048 x 2048), PAD column -> -1e9 ---
  wmma_gemm_nt<<<dim3(V_ / 128, V_ / 64, H_), 256, 0, stream>>>(
      QKVbf, OUT3_, D_,                 // A: s=0, +h*256 per z
      QKVbf + 2048, OUT3_, D_,          // B: s=1, +h*256 per z
      nullptr, nullptr,
      CQK, V_, (long long)V_ * V_, nullptr, D_, invTemp, 1);

  // --- char_pos_qk[h] = (Qc@Kp^T + Kc@Qp^T)/16  (8 x 2048 x 64) ---
  wmma_gemm_nt<<<dim3(V_ / 128, 1, H_), 256, 0, stream>>>(
      QKVbf, OUT3_, D_,                 // A : char Q (s=0)
      PQKVbf + 2048, OUT3_, D_,         // B : pos K  (s=1)
      QKVbf + 2048,                     // A2: char K (s=1)
      PQKVbf,                           // B2: pos Q  (s=0)
      CPQK, MAXLEN_, (long long)V_ * MAXLEN_, nullptr, D_, invTemp, 0);

  // --- pos_qk[h] = Qp @ Kp^T / 16  (8 x 128(pad) x 64) ---
  wmma_gemm_nt<<<dim3(1, 1, H_), 256, 0, stream>>>(
      PQKVbf, OUT3_, D_,
      PQKVbf + 2048, OUT3_, D_,
      nullptr, nullptr,
      PQK, MAXLEN_, (long long)128 * MAXLEN_, nullptr, D_, invTemp, 0);

  // --- per-(word, head) gather + softmax + pooled @ V ---
  k_attn<<<dim3(W_, H_, 1), 64, 0, stream>>>(
      char_code, CQK, CPQK, PQK, QKVf, PQKVf, OUTw, maxlen);

  // --- LayerNorm + per-position MLP head -> y[w, d] ---
  k_head<<<W_, 256, 0, stream>>>(OUTw, ln_gamma, ln_beta, w1, b1, w2, b2, out);
}
